// ConditionalFlowModel_5978594476617
// MI455X (gfx1250) — compile-verified, matrix operand order checked by reading, not run on hardware
//
#include <hip/hip_runtime.h>
#include <hip/hip_bf16.h>
#include <math.h>

// ---------------- problem constants ----------------
#define B_    16
#define L_    2048
#define WIN_  128
#define D_    512
#define H_    8
#define FF_   2048
#define NL_   3
#define V_    32000
#define MM_   32
#define HD_   64
#define S_    129          // WIN+1 (cond token prepended)
#define ROWS_  (B_*S_)     // 2064
#define PROWS_ (B_*WIN_)   // 2048
#define EPS_  1e-5f
#define NEGBIG (-1e30f)

typedef __attribute__((ext_vector_type(16))) __bf16 v16bf;
typedef __attribute__((ext_vector_type(8)))  __bf16 bf16x8;
typedef __attribute__((ext_vector_type(8)))  float  v8f;
typedef __attribute__((ext_vector_type(4)))  unsigned int v4u;
typedef __attribute__((ext_vector_type(4)))  int v4i;
typedef __attribute__((ext_vector_type(8)))  int v8i;

#if __has_builtin(__builtin_amdgcn_tensor_load_to_lds)
#define HAVE_TDM_ 1
#endif

// ---------------- helpers ----------------
static __device__ __forceinline__ unsigned short f2bf(float f) {
  union { float f; unsigned u; } x; x.f = f;
  unsigned u = x.u + 0x7fffu + ((x.u >> 16) & 1u);   // round-to-nearest-even
  return (unsigned short)(u >> 16);
}
static __device__ __forceinline__ float bf2f(unsigned short h) {
  union { unsigned u; float f; } x; x.u = ((unsigned)h) << 16;
  return x.f;
}

// block-wide sum for blockDim.x == 256 (8 waves of 32), broadcast result
static __device__ __forceinline__ float block_sum256(float v) {
  __shared__ float red[8];
  __syncthreads();
  #pragma unroll
  for (int o = 16; o > 0; o >>= 1) v += __shfl_down(v, o, 32);
  int lane = threadIdx.x & 31, w = threadIdx.x >> 5;
  if (lane == 0) red[w] = v;
  __syncthreads();
  if (threadIdx.x == 0) {
    float s = 0.f;
    #pragma unroll
    for (int i = 0; i < 8; ++i) s += red[i];
    red[0] = s;
  }
  __syncthreads();
  return red[0];
}

// Load one 16x32 bf16 WMMA fragment (A layout per ISA 7.12.2; B uses the same
// layout taken from the [N,K] weight matrix). base: bf16 matrix, ldk = K.
static __device__ __forceinline__ v16bf load_frag(const unsigned short* base, int ldk,
                                                  int row0, int kb, int lane) {
  int r = lane & 15, hi = lane >> 4;
  const unsigned short* p = base + (size_t)(row0 + r) * ldk + kb + hi * 8;
  bf16x8 lo = *(const bf16x8*)p;          // K = kb + hi*8 .. +7
  bf16x8 hv = *(const bf16x8*)(p + 16);   // K = kb + 16 + hi*8 .. +7
  return __builtin_shufflevector(lo, hv, 0,1,2,3,4,5,6,7,8,9,10,11,12,13,14,15);
}

// ---------------- elementwise kernels ----------------
__global__ void k_zero(float* p, int n) {
  int i = blockIdx.x * blockDim.x + threadIdx.x;
  if (i < n) p[i] = 0.f;
}

__global__ void k_f32_to_bf16(const float* __restrict__ src, unsigned short* __restrict__ dst, int n) {
  int i = blockIdx.x * blockDim.x + threadIdx.x;
  int stride = gridDim.x * blockDim.x;
  for (; i < n; i += stride) dst[i] = f2bf(src[i]);
}

// ---------------- pooling ----------------
__global__ void k_pools(const int* __restrict__ tok, const int* __restrict__ wsA,
                        const int* __restrict__ weA, const float* __restrict__ emb,
                        float* __restrict__ left, float* __restrict__ right,
                        float* __restrict__ glob) {
  int b = blockIdx.x, tid = threadIdx.x;
  int ws = wsA[b], we = weA[b];
  float l0 = 0, l1 = 0, r0 = 0, r1 = 0, g0 = 0, g1 = 0;
  for (int l = 0; l < L_; ++l) {
    int tk = tok[b * L_ + l];
    const float* e = emb + (size_t)tk * D_;
    float e0 = e[tid], e1 = e[tid + 256];
    g0 += e0; g1 += e1;
    if (l < ws)  { l0 += e0; l1 += e1; }
    if (l >= we) { r0 += e0; r1 += e1; }
  }
  float dl = fmaxf((float)ws, 1.f), dr = fmaxf((float)(L_ - we), 1.f);
  left [b * D_ + tid]       = l0 / dl;  left [b * D_ + tid + 256] = l1 / dl;
  right[b * D_ + tid]       = r0 / dr;  right[b * D_ + tid + 256] = r1 / dr;
  glob [b * D_ + tid]       = g0 / (float)L_;
  glob [b * D_ + tid + 256] = g1 / (float)L_;
}

__global__ void k_motif(const int* __restrict__ mids, const int* __restrict__ mlen,
                        const float* __restrict__ emb, float* __restrict__ motif) {
  int b = blockIdx.x, tid = threadIdx.x;
  int ml = mlen[b];
  for (int d = tid; d < D_; d += 256) {
    float s = 0.f;
    for (int m = 0; m < MM_; ++m)
      if (m < ml) s += emb[(size_t)mids[b * MM_ + m] * D_ + d];
    motif[b * D_ + d] = s / fmaxf((float)ml, 1.f);
  }
}

__global__ void k_condin(const float* __restrict__ motif, const float* __restrict__ left,
                         const float* __restrict__ right, const float* __restrict__ glob,
                         unsigned short* __restrict__ cin) {
  int b = blockIdx.x, tid = threadIdx.x;
  size_t base = (size_t)b * 4 * D_;
  for (int d = tid; d < D_; d += 256) {
    cin[base + d]          = f2bf(motif[b * D_ + d]);
    cin[base + D_ + d]     = f2bf(left [b * D_ + d]);
    cin[base + 2 * D_ + d] = f2bf(right[b * D_ + d]);
    cin[base + 3 * D_ + d] = f2bf(glob [b * D_ + d]);
  }
}

__global__ void k_tin(const float* __restrict__ t, unsigned short* __restrict__ tin) {
  int b = blockIdx.x, j = threadIdx.x;  // 256 threads, half = 256
  float fr = __expf(-logf(10000.f) * (float)j / 255.f);
  float a = t[b] * fr;
  tin[(size_t)b * D_ + j]       = f2bf(__sinf(a));
  tin[(size_t)b * D_ + 256 + j] = f2bf(__cosf(a));
}

// ---------------- generic bf16 WMMA GEMM: C[M,N] = A[M,K] * W[N,K]^T + bias ----------------
// Each wave computes a 16x64 strip (4 accumulators) so every A fragment feeds
// 4 WMMAs. Grid: (N/256, M/16), 128 threads (4 waves) per block.
__global__ void k_gemm_bf16(const unsigned short* __restrict__ A,
                            const unsigned short* __restrict__ W,
                            const float* __restrict__ bias,
                            float* __restrict__ C,
                            unsigned short* __restrict__ Cbf,
                            int Mr, int N, int K, int relu) {
  int tid = threadIdx.x, lane = tid & 31, wv = tid >> 5;
  int n0 = blockIdx.x * 256 + wv * 64;
  int m0 = blockIdx.y * 16;
  if (n0 >= N || m0 >= Mr) return;
  v8f acc0 = {0.f,0.f,0.f,0.f,0.f,0.f,0.f,0.f};
  v8f acc1 = acc0, acc2 = acc0, acc3 = acc0;
  for (int kb = 0; kb < K; kb += 32) {
    v16bf a  = load_frag(A, K, m0, kb, lane);
    v16bf b0 = load_frag(W, K, n0,      kb, lane);
    v16bf b1 = load_frag(W, K, n0 + 16, kb, lane);
    v16bf b2 = load_frag(W, K, n0 + 32, kb, lane);
    v16bf b3 = load_frag(W, K, n0 + 48, kb, lane);
    acc0 = __builtin_amdgcn_wmma_f32_16x16x32_bf16(false, a, false, b0, (short)0, acc0, false, false);
    acc1 = __builtin_amdgcn_wmma_f32_16x16x32_bf16(false, a, false, b1, (short)0, acc1, false, false);
    acc2 = __builtin_amdgcn_wmma_f32_16x16x32_bf16(false, a, false, b2, (short)0, acc2, false, false);
    acc3 = __builtin_amdgcn_wmma_f32_16x16x32_bf16(false, a, false, b3, (short)0, acc3, false, false);
  }
  int rb = m0 + ((lane >> 4) << 3);
  #pragma unroll
  for (int j = 0; j < 4; ++j) {
    v8f acc = (j == 0) ? acc0 : (j == 1) ? acc1 : (j == 2) ? acc2 : acc3;
    int n = n0 + j * 16 + (lane & 15);
    float bv = bias ? bias[n] : 0.f;
    #pragma unroll
    for (int v = 0; v < 8; ++v) {
      float val = acc[v] + bv;
      if (relu) val = fmaxf(val, 0.f);
      if (C)   C  [(size_t)(rb + v) * N + n] = val;
      if (Cbf) Cbf[(size_t)(rb + v) * N + n] = f2bf(val);
    }
  }
}

// ---------------- build x (cond token + x_t + temb), store x1/x_t ----------------
__global__ void k_buildx(const int* __restrict__ tok, const int* __restrict__ wsA,
                         const int* __restrict__ weA, const float* __restrict__ t,
                         const float* __restrict__ x0, const float* __restrict__ emb,
                         const float* __restrict__ cond, const float* __restrict__ temb,
                         float* __restrict__ x, unsigned short* __restrict__ xbf,
                         float* __restrict__ x1buf, float* __restrict__ xtbuf) {
  int b = blockIdx.x, s = blockIdx.y, tid = threadIdx.x;
  size_t xrow = (size_t)(b * S_ + s) * D_;
  if (s == 0) {
    for (int d = tid; d < D_; d += 256) {
      float v = cond[b * D_ + d];
      x[xrow + d] = v; xbf[xrow + d] = f2bf(v);
    }
    return;
  }
  int w = s - 1;
  int ws = wsA[b], we = weA[b];
  int actual = min(max(we - ws, 0), WIN_);
  float tb = t[b];
  int idx = min(ws + w, L_ - 1);
  int tk = tok[b * L_ + idx];
  int valid = (w < actual) ? 1 : 0;
  size_t wrow = (size_t)(b * WIN_ + w) * D_;
  for (int d = tid; d < D_; d += 256) {
    float x1v = valid ? emb[(size_t)tk * D_ + d] : 0.f;
    float xtv = (1.f - tb) * x0[wrow + d] + tb * x1v;
    x1buf[wrow + d] = x1v;
    xtbuf[wrow + d] = xtv;
    float xv = xtv + temb[b * D_ + d];
    x[xrow + d] = xv; xbf[xrow + d] = f2bf(xv);
  }
}

// ---------------- attention (S=129, HD=64): one block per (b,h), online softmax ----------------
__global__ void k_attn(const float* __restrict__ qkv, unsigned short* __restrict__ ctx_bf) {
  __shared__ unsigned short Qs[S_ * HD_], Ks[S_ * HD_], Vs[S_ * HD_];
  int b = blockIdx.x >> 3, h = blockIdx.x & 7;
  int tid = threadIdx.x;
  for (int idx = tid; idx < S_ * HD_; idx += 256) {
    int s = idx / HD_, d = idx % HD_;
    size_t base = (size_t)(b * S_ + s) * (3 * D_) + h * HD_ + d;
    Qs[idx] = f2bf(qkv[base]);
    Ks[idx] = f2bf(qkv[base + D_]);
    Vs[idx] = f2bf(qkv[base + 2 * D_]);
  }
  __syncthreads();
  if (tid >= S_) return;
  float qr[HD_];
  #pragma unroll
  for (int d = 0; d < HD_; ++d) qr[d] = bf2f(Qs[tid * HD_ + d]);
  float acc[HD_];
  #pragma unroll
  for (int d = 0; d < HD_; ++d) acc[d] = 0.f;
  float m = NEGBIG, ssum = 0.f;
  for (int j = 0; j < S_; ++j) {
    float sc = 0.f;
    #pragma unroll 16
    for (int d = 0; d < HD_; ++d) sc += qr[d] * bf2f(Ks[j * HD_ + d]);
    sc *= 0.125f;  // 1/sqrt(64)
    float mn = fmaxf(m, sc);
    float corr = __expf(m - mn);
    float pj = __expf(sc - mn);
    ssum = ssum * corr + pj;
    #pragma unroll 16
    for (int d = 0; d < HD_; ++d) acc[d] = acc[d] * corr + pj * bf2f(Vs[j * HD_ + d]);
    m = mn;
  }
  float inv = 1.f / ssum;
  size_t ob = (size_t)(b * S_ + tid) * D_ + h * HD_;
  #pragma unroll
  for (int d = 0; d < HD_; ++d) ctx_bf[ob + d] = f2bf(acc[d] * inv);
}

// ---------------- LayerNorm(x + delta), writes f32 + bf16 ----------------
__global__ void k_ln(float* __restrict__ x, const float* __restrict__ delta,
                     const float* __restrict__ g, const float* __restrict__ bta,
                     unsigned short* __restrict__ xbf) {
  int row = blockIdx.x, tid = threadIdx.x;
  size_t base = (size_t)row * D_;
  float v0 = x[base + tid] + delta[base + tid];
  float v1 = x[base + tid + 256] + delta[base + tid + 256];
  float mean = block_sum256(v0 + v1) * (1.f / D_);
  float d0 = v0 - mean, d1 = v1 - mean;
  float var = block_sum256(d0 * d0 + d1 * d1) * (1.f / D_);
  float rstd = rsqrtf(var + EPS_);
  float o0 = d0 * rstd * g[tid] + bta[tid];
  float o1 = d1 * rstd * g[tid + 256] + bta[tid + 256];
  x[base + tid] = o0;        xbf[base + tid] = f2bf(o0);
  x[base + tid + 256] = o1;  xbf[base + tid + 256] = f2bf(o1);
}

// ---------------- velocity MSE numerator ----------------
__global__ void k_vel(const float* __restrict__ predv, const float* __restrict__ x1b,
                      const float* __restrict__ x0, const int* __restrict__ wsA,
                      const int* __restrict__ weA, float* __restrict__ accum) {
  int bw = blockIdx.x, b = bw >> 7, w = bw & 127, tid = threadIdx.x;
  int actual = min(max(weA[b] - wsA[b], 0), WIN_);
  if (w >= actual) return;
  size_t prow = (size_t)(b * S_ + 1 + w) * D_;
  size_t wrow = (size_t)(b * WIN_ + w) * D_;
  float lo = 0.f;
  for (int d = tid; d < D_; d += 256) {
    float tv = x1b[wrow + d] - x0[wrow + d];
    float df = predv[prow + d] - tv;
    lo += df * df;
  }
  float s = block_sum256(lo);
  if (tid == 0) atomicAdd(accum, s);
}

// ---------------- pred_x1 = x_t + (1-t)*pred_v -> bf16 ----------------
__global__ void k_predx1(const float* __restrict__ xtb, const float* __restrict__ predv,
                         const float* __restrict__ t, unsigned short* __restrict__ px1bf) {
  int row = blockIdx.x, b = row >> 7, w = row & 127, tid = threadIdx.x;
  float tb = t[b];
  size_t prow = (size_t)(b * S_ + 1 + w) * D_;
  size_t wrow = (size_t)row * D_;
  for (int d = tid; d < D_; d += 256) {
    float px = xtb[wrow + d] + (1.f - tb) * predv[prow + d];
    px1bf[wrow + d] = f2bf(px);
  }
}

// ---------------- fused decode logsumexp partials (WMMA, online softmax) ----------------
// grid (16 slices of 2000 cols, 128 row tiles), block 128 (4 waves).
// The 16x512 bf16 y-tile is staged into LDS by the Tensor Data Mover (D# built
// per ISA 8.3/8.4: 2D tensor 512x2048, tile 512x16, data_size=2B), issued once
// by wave 0; other waves consume after s_wait_tensorcnt + barrier.
__global__ void k_lse_partial(const unsigned short* __restrict__ Ybf,
                              const unsigned short* __restrict__ Wdec,
                              const float* __restrict__ db,
                              float* __restrict__ partials) {
  __shared__ unsigned short Asm[16 * D_];  // 16 KB: 16 rows of y in bf16
  int slice = blockIdx.x;
  int r0 = blockIdx.y * 16;
  int tid = threadIdx.x;
#if defined(HAVE_TDM_)
  if ((tid >> 5) == 0) {   // one TDM issue per block (wave-uniform branch)
    unsigned long long gaddr = (unsigned long long)(const void*)(Ybf + (size_t)r0 * D_);
    unsigned lds_off = (unsigned)(unsigned long long)(void*)&Asm[0];
    v4u g0;
    g0.x = 1u;                                                  // count=1 (valid load D#)
    g0.y = lds_off;                                             // lds_addr (bytes)
    g0.z = (unsigned)(gaddr & 0xFFFFFFFFu);                     // global_addr[31:0]
    g0.w = (unsigned)((gaddr >> 32) & 0x1FFFFFFu) | (2u << 30); // global_addr[56:32] | type=2
    v8i g1;
    g1[0] = (int)(1u << 16);      // data_size=1 -> 2 bytes/elt
    g1[1] = (int)(512u << 16);    // tensor_dim0[15:0]=512 (cols)
    g1[2] = (int)(2048u << 16);   // tensor_dim0 hi=0 | tensor_dim1[15:0]=2048 (rows)
    g1[3] = (int)(512u << 16);    // tensor_dim1 hi=0 | tile_dim0=512
    g1[4] = 16;                   // tile_dim1=16 | tile_dim2=0
    g1[5] = 512;                  // tensor_dim0_stride[31:0]=512 elts
    g1[6] = 0;                    // stride0 hi | stride1 lo (unused for 2D)
    g1[7] = 0;
    v4i zz = {0, 0, 0, 0};
#if __clang_major__ >= 23
    v8i z8 = {0, 0, 0, 0, 0, 0, 0, 0};
    __builtin_amdgcn_tensor_load_to_lds(g0, g1, zz, zz, z8, 0);
#else
    __builtin_amdgcn_tensor_load_to_lds(g0, g1, zz, zz, 0);
#endif
    __builtin_amdgcn_s_wait_tensorcnt(0);
  }
  __syncthreads();
#else
  for (int idx = tid; idx < 16 * D_; idx += 128)
    Asm[idx] = Ybf[(size_t)(r0 + idx / D_) * D_ + (idx % D_)];
  __syncthreads();
#endif
  int lane = tid & 31, wv = tid >> 5;
  int r = lane & 15, hi = lane >> 4;
  float m_run[8], s_run[8];
  #pragma unroll
  for (int v = 0; v < 8; ++v) { m_run[v] = NEGBIG; s_run[v] = 0.f; }
  const int TILES = 125;  // 2000/16
  for (int tt = wv; tt < TILES; tt += 4) {
    int n0 = slice * 2000 + tt * 16;
    if (tt + 4 < TILES)
      __builtin_prefetch(Wdec + (size_t)(n0 + 64 + r) * D_, 0, 1);  // global_prefetch_b8
    v8f acc = {0.f, 0.f, 0.f, 0.f, 0.f, 0.f, 0.f, 0.f};
    #pragma unroll
    for (int kb = 0; kb < D_; kb += 32) {
      v16bf a  = load_frag(Asm, D_, 0, kb, lane);   // ds_load_b128
      v16bf bb = load_frag(Wdec, D_, n0, kb, lane);
      acc = __builtin_amdgcn_wmma_f32_16x16x32_bf16(false, a, false, bb, (short)0, acc, false, false);
    }
    float bn = db[n0 + r];  // this lane's column bias
    #pragma unroll
    for (int v = 0; v < 8; ++v) {
      float val = acc[v] + bn;
      float mn = fmaxf(m_run[v], val);
      s_run[v] = s_run[v] * __expf(m_run[v] - mn) + __expf(val - mn);
      m_run[v] = mn;
    }
  }
  // combine the 16 lanes that hold the same 8 rows
  #pragma unroll
  for (int off = 1; off < 16; off <<= 1) {
    #pragma unroll
    for (int v = 0; v < 8; ++v) {
      float mo = __shfl_xor(m_run[v], off, 32);
      float so = __shfl_xor(s_run[v], off, 32);
      float mn = fmaxf(m_run[v], mo);
      s_run[v] = s_run[v] * __expf(m_run[v] - mn) + so * __expf(mo - mn);
      m_run[v] = mn;
    }
  }
  if (r == 0) {
    int rbase = r0 + hi * 8;
    int pidx = slice * 4 + wv;  // 0..63
    #pragma unroll
    for (int v = 0; v < 8; ++v) {
      partials[((size_t)(rbase + v) * 64 + pidx) * 2 + 0] = m_run[v];
      partials[((size_t)(rbase + v) * 64 + pidx) * 2 + 1] = s_run[v];
    }
  }
}

// ---------------- picked-label logit (f32 dot) ----------------
__global__ void k_picked(const float* __restrict__ y, const float* __restrict__ decW,
                         const float* __restrict__ db, const int* __restrict__ tok,
                         const int* __restrict__ wsA, const int* __restrict__ weA,
                         float* __restrict__ picked) {
  int row = blockIdx.x, b = row >> 7, w = row & 127, tid = threadIdx.x;
  int ws = wsA[b], we = weA[b];
  int actual = min(max(we - ws, 0), WIN_);
  if (w >= actual) { if (tid == 0) picked[row] = 0.f; return; }
  int idx = min(ws + w, L_ - 1);
  int label = tok[b * L_ + idx];
  size_t yrow = (size_t)row * D_;
  size_t wr = (size_t)label * D_;
  float lo = 0.f;
  for (int d = tid; d < D_; d += 256) lo += y[yrow + d] * decW[wr + d];
  float s = block_sum256(lo);
  if (tid == 0) picked[row] = s + db[label];
}

// ---------------- final reduction: loss = vel + 0.1*recon ----------------
__global__ void k_final(const float* __restrict__ partials, const float* __restrict__ picked,
                        const int* __restrict__ wsA, const int* __restrict__ weA,
                        const float* __restrict__ accum, float* __restrict__ out) {
  int tid = threadIdx.x;
  float rs = 0.f, rc = 0.f;
  for (int row = tid; row < PROWS_; row += 256) {
    int b = row >> 7, w = row & 127;
    int actual = min(max(weA[b] - wsA[b], 0), WIN_);
    if (w < actual) {
      float m = NEGBIG, s = 0.f;
      for (int p = 0; p < 64; ++p) {
        float mp = partials[((size_t)row * 64 + p) * 2 + 0];
        float sp = partials[((size_t)row * 64 + p) * 2 + 1];
        float mn = fmaxf(m, mp);
        s = s * __expf(m - mn) + sp * __expf(mp - mn);
        m = mn;
      }
      float lse = m + logf(s);
      rs += (lse - picked[row]);  // -logp
      rc += 1.f;
    }
  }
  rs = block_sum256(rs);
  rc = block_sum256(rc);
  if (tid == 0) {
    float cnt = rc;
    float vel = accum[0] / (fmaxf(cnt * (float)D_, 1.f) * (float)D_);
    float recon = rs / fmaxf(cnt, 1.f);
    out[0] = vel + 0.1f * recon;
  }
}

// ---------------- launcher ----------------
extern "C" void kernel_launch(void* const* d_in, const int* in_sizes, int n_in,
                              void* d_out, int out_size, void* d_ws, size_t ws_size,
                              hipStream_t stream) {
  (void)in_sizes; (void)n_in; (void)out_size; (void)ws_size;
  const int*   token_ids    = (const int*)d_in[0];
  const int*   motif_ids    = (const int*)d_in[1];
  const int*   motif_len    = (const int*)d_in[2];
  const int*   window_start = (const int*)d_in[3];
  const int*   window_end   = (const int*)d_in[4];
  const float* t            = (const float*)d_in[5];
  const float* x0           = (const float*)d_in[6];
  const float* embed        = (const float*)d_in[7];
  const float* decode_W     = (const float*)d_in[8];
  const float* decode_b     = (const float*)d_in[9];
  const float* adapter_W    = (const float*)d_in[10];
  const float* adapter_b    = (const float*)d_in[11];
  const float* time_W       = (const float*)d_in[12];
  const float* time_b       = (const float*)d_in[13];
  const float* cond_W       = (const float*)d_in[14];
  const float* cond_b       = (const float*)d_in[15];
  const float* qkv_W        = (const float*)d_in[16];
  const float* qkv_b        = (const float*)d_in[17];
  const float* ao_W         = (const float*)d_in[18];
  const float* ao_b         = (const float*)d_in[19];
  const float* ff1_W        = (const float*)d_in[20];
  const float* ff1_b        = (const float*)d_in[21];
  const float* ff2_W        = (const float*)d_in[22];
  const float* ff2_b        = (const float*)d_in[23];
  const float* ln1_g        = (const float*)d_in[24];
  const float* ln1_bb       = (const float*)d_in[25];
  const float* ln2_g        = (const float*)d_in[26];
  const float* ln2_bb       = (const float*)d_in[27];
  const float* out_W        = (const float*)d_in[28];
  const float* out_b        = (const float*)d_in[29];
  float* out = (float*)d_out;

  char* p = (char*)d_ws;
  auto alloc = [&](size_t bytes) -> char* {
    char* r = p; p += (bytes + 255) & ~(size_t)255; return r;
  };
  // bf16 weight copies
  unsigned short* wTime = (unsigned short*)alloc((size_t)D_ * D_ * 2);
  unsigned short* wCond = (unsigned short*)alloc((size_t)D_ * 4 * D_ * 2);
  unsigned short* wQkv  = (unsigned short*)alloc((size_t)NL_ * 3 * D_ * D_ * 2);
  unsigned short* wAo   = (unsigned short*)alloc((size_t)NL_ * D_ * D_ * 2);
  unsigned short* wF1   = (unsigned short*)alloc((size_t)NL_ * FF_ * D_ * 2);
  unsigned short* wF2   = (unsigned short*)alloc((size_t)NL_ * D_ * FF_ * 2);
  unsigned short* wOut  = (unsigned short*)alloc((size_t)D_ * D_ * 2);
  unsigned short* wAd   = (unsigned short*)alloc((size_t)D_ * D_ * 2);
  unsigned short* wDec  = (unsigned short*)alloc((size_t)V_ * D_ * 2);
  // activations
  float* poolL  = (float*)alloc((size_t)B_ * D_ * 4);
  float* poolR  = (float*)alloc((size_t)B_ * D_ * 4);
  float* poolG  = (float*)alloc((size_t)B_ * D_ * 4);
  float* poolM  = (float*)alloc((size_t)B_ * D_ * 4);
  float* condv  = (float*)alloc((size_t)B_ * D_ * 4);
  float* temb   = (float*)alloc((size_t)B_ * D_ * 4);
  unsigned short* cin = (unsigned short*)alloc((size_t)B_ * 4 * D_ * 2);
  unsigned short* tin = (unsigned short*)alloc((size_t)B_ * D_ * 2);
  float* x      = (float*)alloc((size_t)ROWS_ * D_ * 4);
  unsigned short* xbf = (unsigned short*)alloc((size_t)ROWS_ * D_ * 2);
  float* qkv    = (float*)alloc((size_t)ROWS_ * 3 * D_ * 4);
  unsigned short* ctxbf = (unsigned short*)alloc((size_t)ROWS_ * D_ * 2);
  float* tmp    = (float*)alloc((size_t)ROWS_ * D_ * 4);
  unsigned short* ffhbf = (unsigned short*)alloc((size_t)ROWS_ * FF_ * 2);
  float* predv  = (float*)alloc((size_t)ROWS_ * D_ * 4);
  float* x1b    = (float*)alloc((size_t)PROWS_ * D_ * 4);
  float* xtb    = (float*)alloc((size_t)PROWS_ * D_ * 4);
  unsigned short* px1bf = (unsigned short*)alloc((size_t)PROWS_ * D_ * 2);
  float* y      = (float*)alloc((size_t)PROWS_ * D_ * 4);
  unsigned short* ybf = (unsigned short*)alloc((size_t)PROWS_ * D_ * 2);
  float* partials = (float*)alloc((size_t)PROWS_ * 64 * 2 * 4);
  float* picked = (float*)alloc((size_t)PROWS_ * 4);
  float* accum  = (float*)alloc(256);

  k_zero<<<1, 64, 0, stream>>>(accum, 64);

  auto conv = [&](const float* s, unsigned short* d, int n) {
    int blocks = (n + 255) / 256; if (blocks > 2048) blocks = 2048;
    k_f32_to_bf16<<<blocks, 256, 0, stream>>>(s, d, n);
  };
  conv(time_W,    wTime, D_ * D_);
  conv(cond_W,    wCond, D_ * 4 * D_);
  conv(qkv_W,     wQkv,  NL_ * 3 * D_ * D_);
  conv(ao_W,      wAo,   NL_ * D_ * D_);
  conv(ff1_W,     wF1,   NL_ * FF_ * D_);
  conv(ff2_W,     wF2,   NL_ * D_ * FF_);
  conv(out_W,     wOut,  D_ * D_);
  conv(adapter_W, wAd,   D_ * D_);
  conv(decode_W,  wDec,  V_ * D_);

  k_pools<<<B_, 256, 0, stream>>>(token_ids, window_start, window_end, embed, poolL, poolR, poolG);
  k_motif<<<B_, 256, 0, stream>>>(motif_ids, motif_len, embed, poolM);
  k_condin<<<B_, 256, 0, stream>>>(poolM, poolL, poolR, poolG, cin);
  k_gemm_bf16<<<dim3(D_ / 256, 1), 128, 0, stream>>>(cin, wCond, cond_b, condv, nullptr, B_, D_, 4 * D_, 0);
  k_tin<<<B_, 256, 0, stream>>>(t, tin);
  k_gemm_bf16<<<dim3(D_ / 256, 1), 128, 0, stream>>>(tin, wTime, time_b, temb, nullptr, B_, D_, D_, 0);
  k_buildx<<<dim3(B_, S_), 256, 0, stream>>>(token_ids, window_start, window_end, t, x0, embed,
                                             condv, temb, x, xbf, x1b, xtb);

  for (int i = 0; i < NL_; ++i) {
    k_gemm_bf16<<<dim3((3 * D_) / 256, ROWS_ / 16), 128, 0, stream>>>(
        xbf, wQkv + (size_t)i * 3 * D_ * D_, qkv_b + i * 3 * D_, qkv, nullptr, ROWS_, 3 * D_, D_, 0);
    k_attn<<<B_ * H_, 256, 0, stream>>>(qkv, ctxbf);
    k_gemm_bf16<<<dim3(D_ / 256, ROWS_ / 16), 128, 0, stream>>>(
        ctxbf, wAo + (size_t)i * D_ * D_, ao_b + i * D_, tmp, nullptr, ROWS_, D_, D_, 0);
    k_ln<<<ROWS_, 256, 0, stream>>>(x, tmp, ln1_g + i * D_, ln1_bb + i * D_, xbf);
    k_gemm_bf16<<<dim3(FF_ / 256, ROWS_ / 16), 128, 0, stream>>>(
        xbf, wF1 + (size_t)i * FF_ * D_, ff1_b + i * FF_, nullptr, ffhbf, ROWS_, FF_, D_, 1);
    k_gemm_bf16<<<dim3(D_ / 256, ROWS_ / 16), 128, 0, stream>>>(
        ffhbf, wF2 + (size_t)i * D_ * FF_, ff2_b + i * D_, tmp, nullptr, ROWS_, D_, FF_, 0);
    k_ln<<<ROWS_, 256, 0, stream>>>(x, tmp, ln2_g + i * D_, ln2_bb + i * D_, xbf);
  }

  k_gemm_bf16<<<dim3(D_ / 256, ROWS_ / 16), 128, 0, stream>>>(
      xbf, wOut, out_b, predv, nullptr, ROWS_, D_, D_, 0);
  k_vel<<<PROWS_, 256, 0, stream>>>(predv, x1b, x0, window_start, window_end, accum);
  k_predx1<<<PROWS_, 256, 0, stream>>>(xtb, predv, t, px1bf);
  k_gemm_bf16<<<dim3(D_ / 256, PROWS_ / 16), 128, 0, stream>>>(
      px1bf, wAd, adapter_b, y, ybf, PROWS_, D_, D_, 0);
  k_lse_partial<<<dim3(16, PROWS_ / 16), 128, 0, stream>>>(ybf, wDec, decode_b, partials);
  k_picked<<<PROWS_, 256, 0, stream>>>(y, decode_W, decode_b, token_ids, window_start, window_end, picked);
  k_final<<<1, 256, 0, stream>>>(partials, picked, window_start, window_end, accum, out);
}
